// RandLANetRes_10479720202825
// MI455X (gfx1250) — matrix operand order repaired and checked by reading, not compiled, not run on hardware
//
#include <hip/hip_runtime.h>
#include <math.h>

typedef __attribute__((ext_vector_type(16))) _Float16 v16h;
typedef __attribute__((ext_vector_type(8)))  _Float16 v8h;
typedef __attribute__((ext_vector_type(4)))  _Float16 v4h;
typedef __attribute__((ext_vector_type(2)))  _Float16 v2h;
typedef __attribute__((ext_vector_type(8)))  float    v8f;
typedef __attribute__((ext_vector_type(4)))  int      v4i;

#define LDF 104   // f16 row stride for 96-wide LDS tiles (208B = 13*16B)
#define LDX 72    // f16 row stride for 64-wide LDS tiles (144B = 9*16B)

#if __has_builtin(__builtin_amdgcn_global_load_async_to_lds_b128) && \
    __has_builtin(__builtin_amdgcn_s_wait_asynccnt)
#define USE_ASYNC_GATHER 1
typedef __attribute__((address_space(1))) v4i* gptr_v4i;
typedef __attribute__((address_space(3))) v4i* lptr_v4i;
#endif

// ---------------------------------------------------------------------------
// WMMA helpers (CDNA5 wave32, V_WMMA_F32_16X16X32_F16)
// ---------------------------------------------------------------------------
__device__ __forceinline__ v8f wmma_f16(v16h a, v16h b, v8f c) {
  return __builtin_amdgcn_wmma_f32_16x16x32_f16(false, a, false, b, (short)0, c,
                                                false, false);
}

__device__ __forceinline__ v8f bcast8(float v) {
  v8f r = {v, v, v, v, v, v, v, v};
  return r;
}

__device__ __forceinline__ v16h cat8(v8h lo, v8h hi) {
  return __builtin_shufflevector(lo, hi, 0, 1, 2, 3, 4, 5, 6, 7,
                                         8, 9, 10, 11, 12, 13, 14, 15);
}

// A fragment: 16x32 f16 tile, row-major LDS (leading dim ld halves), K offset K0.
__device__ __forceinline__ v16h a_frag(const _Float16* A, int ld, int K0) {
  const int l   = threadIdx.x & 31;
  const int row = l & 15;
  const int kb  = (l < 16) ? 0 : 8;
  const v8h* p = (const v8h*)(A + row * ld + K0 + kb);
  return cat8(p[0], p[2]);   // +0 and +16 halves
}

// B fragment from pre-swizzled weights: layout [frag][plane(2)][lane(32)][8 halves].
__device__ __forceinline__ v16h b_frag_sw(const _Float16* Wsw, int f) {
  const int l = threadIdx.x & 31;
  const v8h* base = (const v8h*)(Wsw + f * 512);
  return cat8(base[l], base[32 + l]);
}

// Store 16x16 f32 accumulator (lane=col, VGPR r = rows r/r+8) as ReLU f16 tile.
__device__ __forceinline__ void store_acc_relu(_Float16* dst, int ld, int N0, v8f acc) {
  const int l   = threadIdx.x & 31;
  const int col = l & 15;
  const int mb  = (l < 16) ? 0 : 8;
#pragma unroll
  for (int r = 0; r < 8; r += 2) {
    v2h h;
    h[0] = (_Float16)acc[r];
    h[1] = (_Float16)acc[r + 1];
    v2h z = {(_Float16)0.f, (_Float16)0.f};
    v2h m = __builtin_elementwise_max(h, z);
    dst[(mb + r)     * ld + N0 + col] = m[0];
    dst[(mb + r + 1) * ld + N0 + col] = m[1];
  }
}

// ---------------------------------------------------------------------------
// Weight pre-swizzle: W[K][Nout] f32 -> fragment-ordered f16.
// ---------------------------------------------------------------------------
__global__ void swizzle_weight_kernel(const float* __restrict__ W,
                                      _Float16* __restrict__ dst,
                                      int Kdim, int Nout, int nKc, int nNt)
{
  int total = nKc * nNt * 512;
  int idx = blockIdx.x * blockDim.x + threadIdx.x;
  if (idx >= total) return;
  int j     = idx & 7;
  int lane  = (idx >> 3) & 31;
  int plane = (idx >> 8) & 1;
  int f     = idx >> 9;
  int n  = f % nNt;
  int kc = f / nNt;
  int K = kc * 32 + ((lane < 16) ? 0 : 8) + plane * 16 + j;
  int N = n * 16 + (lane & 15);
  float v = (K < Kdim) ? W[K * Nout + N] : 0.f;
  dst[idx] = (_Float16)v;
}

__global__ void f32_to_f16_kernel(const float* __restrict__ in,
                                  _Float16* __restrict__ out, int n)
{
  int i = blockIdx.x * blockDim.x + threadIdx.x;
  if (i < n) out[i] = (_Float16)in[i];
}

__global__ void zero_kernel(float* __restrict__ p, int n) {
  int i = blockIdx.x * blockDim.x + threadIdx.x;
  if (i < n) p[i] = 0.f;
}

// ---------------------------------------------------------------------------
// Edge kernel: one wave per 16-edge tile. x is f16 [N,64].
// Swizzled-weight block (20480 halves): P1s(1024) P2s(1024) A1s(9216) A2s(9216)
// ---------------------------------------------------------------------------
struct __align__(16) EdgeLds {
  _Float16 w[20480];
  _Float16 fij[2][16 * LDF];
  _Float16 hbuf[2][16 * LDF];
  _Float16 gbuf[2][16 * LDF];
};

__launch_bounds__(64)
__global__ void randla_edge_kernel(
    const _Float16* __restrict__ xh, const float* __restrict__ pos,
    const int* __restrict__ srcIdx, const int* __restrict__ dstIdx,
    const _Float16* __restrict__ swW,
    const float* __restrict__ bp1, const float* __restrict__ bp2,
    const float* __restrict__ ba1, const float* __restrict__ ba2,
    float* __restrict__ aggr, int nTiles, int nE)
{
  __shared__ EdgeLds lds;
  {
    const int4* s = (const int4*)swW;
    int4* d = (int4*)lds.w;
    for (int i = threadIdx.x; i < 20480 / 8; i += blockDim.x) d[i] = s[i];
  }
  __syncthreads();
  const _Float16* wP1s = lds.w;
  const _Float16* wP2s = lds.w + 1024;
  const _Float16* wA1s = lds.w + 2048;
  const _Float16* wA2s = lds.w + 11264;

  const int l   = threadIdx.x & 31;
  const int wv  = threadIdx.x >> 5;
  const int col = l & 15;
  const int row = l & 15;
  _Float16* fij = lds.fij[wv];
  _Float16* hb  = lds.hbuf[wv];
  _Float16* gb  = lds.gbuf[wv];

  float bvp1[2], bvp2[2], bva1[6], bva2[6];
#pragma unroll
  for (int n = 0; n < 2; ++n) { bvp1[n] = bp1[n * 16 + col]; bvp2[n] = bp2[n * 16 + col]; }
#pragma unroll
  for (int n = 0; n < 6; ++n) { bva1[n] = ba1[n * 16 + col]; bva2[n] = ba2[n * 16 + col]; }

  const int waveStride = gridDim.x * 2;
  for (int tile = blockIdx.x * 2 + wv; tile < nTiles; tile += waveStride) {
    int e = tile * 16 + row;
    if (e >= nE) e = nE - 1;
    const int sj = srcIdx[e];
    const int di = dstIdx[e];

    // ---- gather x_j (f16) -> fij[:, 0:64]  (async to LDS when available;
    //      latency hides behind the pp-MLP WMMAs below) ----
    {
      const int cb = (l < 16) ? 0 : 32;
      const _Float16* gsrc = xh + (size_t)sj * 64 + cb;
      _Float16* ldst = fij + row * LDF + cb;
#ifdef USE_ASYNC_GATHER
#pragma unroll
      for (int q = 0; q < 4; ++q) {
        __builtin_amdgcn_global_load_async_to_lds_b128(
            (gptr_v4i)(gsrc + q * 8),
            (lptr_v4i)(ldst + q * 8),
            0, 0);
      }
#else
      const v8h* xr = (const v8h*)gsrc;
      v8h* dd = (v8h*)ldst;
#pragma unroll
      for (int q = 0; q < 4; ++q) dd[q] = xr[q];
#endif
    }

    // ---- relative position encoding ----
    float pix = pos[di * 3 + 0], piy = pos[di * 3 + 1], piz = pos[di * 3 + 2];
    float pjx = pos[sj * 3 + 0], pjy = pos[sj * 3 + 1], pjz = pos[sj * 3 + 2];
    float vx = pix - pjx, vy = piy - pjy, vz = piz - pjz;
    float dij = sqrtf(vx * vx + vy * vy + vz * vz);
    float rel[10] = {pix, piy, piz, pjx, pjy, pjz, vx, vy, vz, dij};

    // A fragment for pp layer 1 (16x32, cols 10..31 zero)
    v16h ap;
#pragma unroll
    for (int j = 0; j < 16; ++j) ap[j] = (_Float16)0.f;
    if (l < 16) {
#pragma unroll
      for (int j = 0; j < 8; ++j) ap[j] = (_Float16)rel[j];
    } else {
      ap[0] = (_Float16)rel[8];
      ap[1] = (_Float16)rel[9];
    }

    // ---- pp layer 1: (10->)32 -> 32  (batched WMMAs, then stores) ----
    {
      v8f acc[2];
#pragma unroll
      for (int n = 0; n < 2; ++n) acc[n] = bcast8(bvp1[n]);
#pragma unroll
      for (int n = 0; n < 2; ++n) acc[n] = wmma_f16(ap, b_frag_sw(wP1s, n), acc[n]);
#pragma unroll
      for (int n = 0; n < 2; ++n) store_acc_relu(hb, LDF, n * 16, acc[n]);
    }
    __builtin_amdgcn_wave_barrier();

    // ---- pp layer 2: 32 -> 32  -> fij[:, 64:96] ----
    {
      v16h a2 = a_frag(hb, LDF, 0);
      v8f acc[2];
#pragma unroll
      for (int n = 0; n < 2; ++n) acc[n] = bcast8(bvp2[n]);
#pragma unroll
      for (int n = 0; n < 2; ++n) acc[n] = wmma_f16(a2, b_frag_sw(wP2s, n), acc[n]);
#pragma unroll
      for (int n = 0; n < 2; ++n) store_acc_relu(fij, LDF, 64 + n * 16, acc[n]);
    }
#ifdef USE_ASYNC_GATHER
    __builtin_amdgcn_s_wait_asynccnt(0);   // x_j gather must have landed in fij
#endif
    __builtin_amdgcn_wave_barrier();

    // ---- attn layer 1: 96 -> 96 (18 back-to-back WMMAs, then stores) ----
    {
      v16h a0 = a_frag(fij, LDF, 0);
      v16h a1 = a_frag(fij, LDF, 32);
      v16h a2 = a_frag(fij, LDF, 64);
      v8f acc[6];
#pragma unroll
      for (int n = 0; n < 6; ++n) acc[n] = bcast8(bva1[n]);
#pragma unroll
      for (int n = 0; n < 6; ++n) {
        acc[n] = wmma_f16(a0, b_frag_sw(wA1s, n),      acc[n]);
        acc[n] = wmma_f16(a1, b_frag_sw(wA1s, 6 + n),  acc[n]);
        acc[n] = wmma_f16(a2, b_frag_sw(wA1s, 12 + n), acc[n]);
      }
#pragma unroll
      for (int n = 0; n < 6; ++n) store_acc_relu(hb, LDF, n * 16, acc[n]);
    }
    __builtin_amdgcn_wave_barrier();

    // ---- attn layer 2: 96 -> 96 ----
    {
      v16h a0 = a_frag(hb, LDF, 0);
      v16h a1 = a_frag(hb, LDF, 32);
      v16h a2 = a_frag(hb, LDF, 64);
      v8f acc[6];
#pragma unroll
      for (int n = 0; n < 6; ++n) acc[n] = bcast8(bva2[n]);
#pragma unroll
      for (int n = 0; n < 6; ++n) {
        acc[n] = wmma_f16(a0, b_frag_sw(wA2s, n),      acc[n]);
        acc[n] = wmma_f16(a1, b_frag_sw(wA2s, 6 + n),  acc[n]);
        acc[n] = wmma_f16(a2, b_frag_sw(wA2s, 12 + n), acc[n]);
      }
#pragma unroll
      for (int n = 0; n < 6; ++n) store_acc_relu(gb, LDF, n * 16, acc[n]);
    }
    __builtin_amdgcn_wave_barrier();

    // ---- feature-dim softmax + msg + scatter-add ----
    const int cb = (l < 16) ? 0 : 48;
    const _Float16* grow = gb  + row * LDF + cb;
    const _Float16* frow = fij + row * LDF + cb;
    _Float16*       erow = hb  + row * LDF + cb;   // hbuf is dead: stage exp here

    v8h g8[6];
#pragma unroll
    for (int c8 = 0; c8 < 6; ++c8) g8[c8] = *(const v8h*)(grow + c8 * 8);

    v8h mv = g8[0];
#pragma unroll
    for (int c8 = 1; c8 < 6; ++c8) mv = __builtin_elementwise_max(mv, g8[c8]);
    v4h m4 = __builtin_elementwise_max(
        __builtin_shufflevector(mv, mv, 0, 1, 2, 3),
        __builtin_shufflevector(mv, mv, 4, 5, 6, 7));
    v2h m2 = __builtin_elementwise_max(
        __builtin_shufflevector(m4, m4, 0, 1),
        __builtin_shufflevector(m4, m4, 2, 3));
    _Float16 mh = (m2[0] > m2[1]) ? m2[0] : m2[1];
    float m = (float)mh;
    m = fmaxf(m, __shfl_xor(m, 16, 32));

    float sum = 0.f;
#pragma unroll
    for (int c8 = 0; c8 < 6; ++c8) {
      float e0 = __expf((float)g8[c8][0] - m);
      float e1 = __expf((float)g8[c8][1] - m);
      float e2 = __expf((float)g8[c8][2] - m);
      float e3 = __expf((float)g8[c8][3] - m);
      float e4 = __expf((float)g8[c8][4] - m);
      float e5 = __expf((float)g8[c8][5] - m);
      float e6 = __expf((float)g8[c8][6] - m);
      float e7 = __expf((float)g8[c8][7] - m);
      sum += ((e0 + e1) + (e2 + e3)) + ((e4 + e5) + (e6 + e7));
      v8h e8;
      e8[0] = (_Float16)e0; e8[1] = (_Float16)e1; e8[2] = (_Float16)e2; e8[3] = (_Float16)e3;
      e8[4] = (_Float16)e4; e8[5] = (_Float16)e5; e8[6] = (_Float16)e6; e8[7] = (_Float16)e7;
      *(v8h*)(erow + c8 * 8) = e8;               // e in [0,1]: safe in f16
    }
    sum += __shfl_xor(sum, 16, 32);
    float inv = 1.f / sum;
    __builtin_amdgcn_wave_barrier();

    float* arow = aggr + (size_t)di * 96 + cb;
#pragma unroll
    for (int c8 = 0; c8 < 6; ++c8) {
      v8h e8 = *(const v8h*)(erow + c8 * 8);
      v8h f8 = *(const v8h*)(frow + c8 * 8);
#pragma unroll
      for (int j = 0; j < 8; ++j) {
        float s = (float)e8[j] * inv;
        atomicAdd(arow + c8 * 8 + j, s * (float)f8[j]);
      }
    }
    __builtin_amdgcn_wave_barrier();
  }
}

// ---------------------------------------------------------------------------
// gn1: h1 = relu(aggr1 @ Wg[96,64] + bg), output in f16 (feeds layer-2 gather)
// ---------------------------------------------------------------------------
struct __align__(16) Gn64Lds {
  _Float16 w[6144];
  _Float16 ta[2][16 * LDF];
};

__launch_bounds__(64)
__global__ void gn64_kernel(const float* __restrict__ aggr,
                            const _Float16* __restrict__ swW,
                            const float* __restrict__ bg,
                            _Float16* __restrict__ out, int nTiles, int nRows)
{
  __shared__ Gn64Lds lds;
  {
    const int4* s = (const int4*)swW;
    int4* d = (int4*)lds.w;
    for (int i = threadIdx.x; i < 6144 / 8; i += blockDim.x) d[i] = s[i];
  }
  __syncthreads();

  const int l = threadIdx.x & 31, wv = threadIdx.x >> 5;
  const int col = l & 15, row = l & 15, mb = (l < 16) ? 0 : 8;
  _Float16* ta = lds.ta[wv];

  const int tile = blockIdx.x * 2 + wv;
  if (tile >= nTiles) return;
  const int g0 = tile * 16;

  {
    int srow = g0 + row; if (srow >= nRows) srow = nRows - 1;
    const int cb = (l < 16) ? 0 : 48;
    const float4* sp = (const float4*)(aggr + (size_t)srow * 96 + cb);
#pragma unroll
    for (int q = 0; q < 12; ++q) {
      float4 v4 = sp[q];
      _Float16* dd = ta + row * LDF + cb + q * 4;
      dd[0] = (_Float16)v4.x; dd[1] = (_Float16)v4.y;
      dd[2] = (_Float16)v4.z; dd[3] = (_Float16)v4.w;
    }
  }
  __builtin_amdgcn_wave_barrier();

  v16h a0 = a_frag(ta, LDF, 0), a1 = a_frag(ta, LDF, 32), a2 = a_frag(ta, LDF, 64);
  v8f acc[4];
#pragma unroll
  for (int n = 0; n < 4; ++n) acc[n] = bcast8(bg[n * 16 + col]);
#pragma unroll
  for (int n = 0; n < 4; ++n) {
    acc[n] = wmma_f16(a0, b_frag_sw(lds.w, n),     acc[n]);
    acc[n] = wmma_f16(a1, b_frag_sw(lds.w, 4 + n), acc[n]);
    acc[n] = wmma_f16(a2, b_frag_sw(lds.w, 8 + n), acc[n]);
  }
#pragma unroll
  for (int n = 0; n < 4; ++n) {
#pragma unroll
    for (int r = 0; r < 8; ++r) {
      int orow = g0 + mb + r;
      if (orow < nRows) {
        float v = fmaxf(acc[n][r], 0.f);
        out[(size_t)orow * 64 + n * 16 + col] = (_Float16)v;
      }
    }
  }
}

// ---------------------------------------------------------------------------
// Final: out = relu( relu(aggr2 @ Wg[96,128] + bg) + x @ Wsc[64,128] + bsc )
// ---------------------------------------------------------------------------
struct __align__(16) FinalLds {
  _Float16 w[20480];
  _Float16 ta[2][16 * LDF];
  _Float16 tx[2][16 * LDX];
};

__launch_bounds__(64)
__global__ void gn128_res_kernel(const float* __restrict__ aggr,
                                 const _Float16* __restrict__ xh,
                                 const _Float16* __restrict__ swW,
                                 const float* __restrict__ bg,
                                 const float* __restrict__ bsc,
                                 float* __restrict__ out, int nTiles, int nRows)
{
  __shared__ FinalLds lds;
  {
    const int4* s = (const int4*)swW;
    int4* d = (int4*)lds.w;
    for (int i = threadIdx.x; i < 20480 / 8; i += blockDim.x) d[i] = s[i];
  }
  __syncthreads();
  const _Float16* wg = lds.w;
  const _Float16* ws = lds.w + 12288;

  const int l = threadIdx.x & 31, wv = threadIdx.x >> 5;
  const int col = l & 15, row = l & 15, mb = (l < 16) ? 0 : 8;
  _Float16* ta = lds.ta[wv];
  _Float16* tx = lds.tx[wv];

  const int tile = blockIdx.x * 2 + wv;
  if (tile >= nTiles) return;
  const int g0 = tile * 16;

  {
    int srow = g0 + row; if (srow >= nRows) srow = nRows - 1;
    const int cb = (l < 16) ? 0 : 48;
    const float4* sp = (const float4*)(aggr + (size_t)srow * 96 + cb);
#pragma unroll
    for (int q = 0; q < 12; ++q) {
      float4 v4 = sp[q];
      _Float16* dd = ta + row * LDF + cb + q * 4;
      dd[0] = (_Float16)v4.x; dd[1] = (_Float16)v4.y;
      dd[2] = (_Float16)v4.z; dd[3] = (_Float16)v4.w;
    }
    const int cb2 = (l < 16) ? 0 : 32;
    const v8h* xp = (const v8h*)(xh + (size_t)srow * 64 + cb2);
    v8h* dd2 = (v8h*)(tx + row * LDX + cb2);
#pragma unroll
    for (int q = 0; q < 4; ++q) dd2[q] = xp[q];
  }
  __builtin_amdgcn_wave_barrier();

  v16h a0 = a_frag(ta, LDF, 0), a1 = a_frag(ta, LDF, 32), a2 = a_frag(ta, LDF, 64);
  v16h x0 = a_frag(tx, LDX, 0), x1 = a_frag(tx, LDX, 32);
#pragma unroll
  for (int g = 0; g < 2; ++g) {
    v8f accA[4], accB[4];
#pragma unroll
    for (int k = 0; k < 4; ++k) {
      int n = g * 4 + k;
      accA[k] = bcast8(bg[n * 16 + col]);
      accB[k] = bcast8(bsc[n * 16 + col]);
    }
#pragma unroll
    for (int k = 0; k < 4; ++k) {
      int n = g * 4 + k;
      accA[k] = wmma_f16(a0, b_frag_sw(wg, n),      accA[k]);
      accA[k] = wmma_f16(a1, b_frag_sw(wg, 8 + n),  accA[k]);
      accA[k] = wmma_f16(a2, b_frag_sw(wg, 16 + n), accA[k]);
      accB[k] = wmma_f16(x0, b_frag_sw(ws, n),      accB[k]);
      accB[k] = wmma_f16(x1, b_frag_sw(ws, 8 + n),  accB[k]);
    }
#pragma unroll
    for (int k = 0; k < 4; ++k) {
      int n = g * 4 + k;
#pragma unroll
      for (int r = 0; r < 8; ++r) {
        int orow = g0 + mb + r;
        if (orow < nRows) {
          float h2 = fmaxf(accA[k][r], 0.f);       // inner ReLU on h2
          float v = fmaxf(h2 + accB[k][r], 0.f);   // residual ReLU
          out[(size_t)orow * 128 + n * 16 + col] = v;
        }
      }
    }
  }
}

// ---------------------------------------------------------------------------
// Launch
// ---------------------------------------------------------------------------
extern "C" void kernel_launch(void* const* d_in, const int* in_sizes, int n_in,
                              void* d_out, int out_size, void* d_ws, size_t ws_size,
                              hipStream_t stream)
{
  const float* x     = (const float*)d_in[0];   // [N,64]
  const float* pos   = (const float*)d_in[1];   // [N,3]
  const int*   ei1   = (const int*)  d_in[2];   // [2,E]
  const int*   ei2   = (const int*)  d_in[3];   // [2,E]
  const float* p1W1  = (const float*)d_in[4];
  const float* p1b1  = (const float*)d_in[5];
  const float* p1W2  = (const float*)d_in[6];
  const float* p1b2  = (const float*)d_in[7];
  const float* a1W1  = (const float*)d_in[8];
  const float* a1b1  = (const float*)d_in[9];
  const float* a1W2  = (const float*)d_in[10];
  const float* a1b2  = (const float*)d_in[11];
  const float* g1W   = (const float*)d_in[12];
  const float* g1b   = (const float*)d_in[13];
  const float* p2W1  = (const float*)d_in[14];
  const float* p2b1  = (const float*)d_in[15];
  const float* p2W2  = (const float*)d_in[16];
  const float* p2b2  = (const float*)d_in[17];
  const float* a2W1  = (const float*)d_in[18];
  const float* a2b1  = (const float*)d_in[19];
  const float* a2W2  = (const float*)d_in[20];
  const float* a2b2  = (const float*)d_in[21];
  const float* g2W   = (const float*)d_in[22];
  const float* g2b   = (const float*)d_in[23];
  const float* scW   = (const float*)d_in[24];
  const float* scb   = (const float*)d_in[25];

  const int N = in_sizes[0] / 64;
  const int E = in_sizes[2] / 2;
  const int nTilesE = (E + 15) / 16;
  const int nTilesN = (N + 15) / 16;

  float*    aggr1 = (float*)d_ws;                         // [N,96] f32
  float*    aggr2 = aggr1 + (size_t)N * 96;               // [N,96] f32
  _Float16* xh    = (_Float16*)(aggr2 + (size_t)N * 96);  // [N,64] f16
  _Float16* h1h   = xh + (size_t)N * 64;                  // [N,64] f16
  _Float16* sw    = h1h + (size_t)N * 64;                 // 67584 halves
  _Float16* swL1  = sw;                 // P1(1024) P2(1024) A1(9216) A2(9216)
  _Float16* swL2  = sw + 20480;
  _Float16* swG1  = sw + 40960;         // 6144
  _Float16* swGF  = sw + 47104;         // wg2(12288) + wsc(8192)
  float*    outp  = (float*)d_out;      // [N,128]

  f32_to_f16_kernel<<<(N * 64 + 255) / 256, 256, 0, stream>>>(x, xh, N * 64);
  {
    dim3 b(256);
    swizzle_weight_kernel<<<(1024 + 255) / 256, b, 0, stream>>>(p1W1, swL1,          10, 32, 1, 2);
    swizzle_weight_kernel<<<(1024 + 255) / 256, b, 0, stream>>>(p1W2, swL1 + 1024,   32, 32, 1, 2);
    swizzle_weight_kernel<<<(9216 + 255) / 256, b, 0, stream>>>(a1W1, swL1 + 2048,   96, 96, 3, 6);
    swizzle_weight_kernel<<<(9216 + 255) / 256, b, 0, stream>>>(a1W2, swL1 + 11264,  96, 96, 3, 6);
    swizzle_weight_kernel<<<(1024 + 255) / 256, b, 0, stream>>>(p2W1, swL2,          10, 32, 1, 2);
    swizzle_weight_kernel<<<(1024 + 255) / 256, b, 0, stream>>>(p2W2, swL2 + 1024,   32, 32, 1, 2);
    swizzle_weight_kernel<<<(9216 + 255) / 256, b, 0, stream>>>(a2W1, swL2 + 2048,   96, 96, 3, 6);
    swizzle_weight_kernel<<<(9216 + 255) / 256, b, 0, stream>>>(a2W2, swL2 + 11264,  96, 96, 3, 6);
    swizzle_weight_kernel<<<(6144 + 255) / 256, b, 0, stream>>>(g1W,  swG1,          96, 64, 3, 4);
    swizzle_weight_kernel<<<(12288 + 255) / 256, b, 0, stream>>>(g2W, swGF,          96, 128, 3, 8);
    swizzle_weight_kernel<<<(8192 + 255) / 256, b, 0, stream>>>(scW,  swGF + 12288,  64, 128, 2, 8);
  }

  {
    int total = N * 96;
    int zb = (total + 255) / 256;
    zero_kernel<<<zb, 256, 0, stream>>>(aggr1, total);
    zero_kernel<<<zb, 256, 0, stream>>>(aggr2, total);
  }

  int edgeBlocks = 1280;
  if (edgeBlocks * 2 > nTilesE) edgeBlocks = (nTilesE + 1) / 2;

  // Layer 1
  randla_edge_kernel<<<edgeBlocks, 64, 0, stream>>>(
      xh, pos, ei1, ei1 + E, swL1, p1b1, p1b2, a1b1, a1b2, aggr1, nTilesE, E);
  gn64_kernel<<<(nTilesN + 1) / 2, 64, 0, stream>>>(aggr1, swG1, g1b, h1h, nTilesN, N);

  // Layer 2 (input = h1 in f16)
  randla_edge_kernel<<<edgeBlocks, 64, 0, stream>>>(
      h1h, pos, ei2, ei2 + E, swL2, p2b1, p2b2, a2b1, a2b2, aggr2, nTilesE, E);

  // gn2 + shortcut + residual ReLU
  gn128_res_kernel<<<(nTilesN + 1) / 2, 64, 0, stream>>>(
      aggr2, xh, swGF, g2b, scb, outp, nTilesN, N);
}
